// NN_47820165874131
// MI455X (gfx1250) — matrix operand order repaired and checked by reading, not compiled, not run on hardware
//
#include <hip/hip_runtime.h>
#include <hip/hip_bf16.h>
#include <math.h>
#include <stdint.h>

// ---------------------------------------------------------------------------
// Problem constants (from reference)
// ---------------------------------------------------------------------------
#define BN_   4
#define CIN_  5
#define HF_   64
#define WF_   2048
#define HWF_  (HF_*WF_)      // 131072
#define HH_   32
#define WH_   1024
#define HWH_  (HH_*WH_)      // 32768

typedef __attribute__((ext_vector_type(16))) _Float16 v16h;
typedef __attribute__((ext_vector_type(8)))  _Float16 v8h;
typedef __attribute__((ext_vector_type(8)))  float    v8f;
typedef __attribute__((ext_vector_type(4)))  unsigned v4u;
typedef __attribute__((ext_vector_type(8)))  int      v8i;
typedef __attribute__((ext_vector_type(4)))  int      v4i;

// ---------------------------------------------------------------------------
// Weight repack: OIHW f32 -> f16 [co][K], K ordered (r,s,c) with c fastest so
// NHWC im2col runs are contiguous.  dst[o][(r*KW+s)*ci + c] = src[o][c][r][s]
// ---------------------------------------------------------------------------
__global__ void pack_weight_kernel(const float* __restrict__ src,
                                   _Float16* __restrict__ dst,
                                   int ci, int khw, int n) {
    int idx = blockIdx.x * blockDim.x + threadIdx.x;
    if (idx >= n) return;
    int ck = ci * khw;
    int o  = idx / ck;
    int rem = idx - o * ck;
    int rs = rem / ci;
    int c  = rem - rs * ci;
    dst[idx] = (_Float16)src[(size_t)(o * ci + c) * khw + rs];
}

// ---------------------------------------------------------------------------
// TDM: DMA one A-tile (rows x 32 f16, row stride Kdim elems) into LDS.
// D# per CDNA5 ISA ch.8: group0 = {count=1, lds_addr, global_addr, type=2},
// group1 = {data_size=2B, tensor_dim0=Kdim, tensor_dim1=rows, tile=32 x rows,
//           dim0_stride=Kdim}. Groups 2/3 zero (2D tensor). TENSORcnt-tracked.
// ---------------------------------------------------------------------------
__device__ __forceinline__ void tdm_load_a_tile(const _Float16* gsrc,
                                                unsigned lds_byte_off,
                                                int Kdim, int rows) {
    unsigned long long ga = (unsigned long long)(uintptr_t)gsrc;
    v4u g0;
    g0.x = 1u;                                            // count=1, user mode
    g0.y = lds_byte_off;                                  // LDS address (bytes)
    g0.z = (unsigned)(ga & 0xffffffffu);                  // global addr lo
    g0.w = (unsigned)((ga >> 32) & 0x01ffffffu) | (2u << 30); // addr hi | type=2
    v8i g1;
    g1[0] = 0x00010000;                                   // data_size = 1 (2 bytes)
    g1[1] = (int)(((unsigned)Kdim) << 16);                // tensor_dim0[15:0]
    g1[2] = (int)((((unsigned)Kdim) >> 16) |
                  (((unsigned)rows) << 16));              // dim0[31:16] | dim1[15:0]
    g1[3] = (int)(32u << 16);                             // dim1[31:16]=0 | tile_dim0=32
    g1[4] = (int)(unsigned)rows;                          // tile_dim1 | tile_dim2=0
    g1[5] = (int)(unsigned)Kdim;                          // dim0_stride lo32
    g1[6] = 0;
    g1[7] = 0;
    v4i z4 = {0, 0, 0, 0};
#if __clang_major__ >= 23
    v8i z8 = {0, 0, 0, 0, 0, 0, 0, 0};
    __builtin_amdgcn_tensor_load_to_lds(g0, g1, z4, z4, z8, 0);
#else
    __builtin_amdgcn_tensor_load_to_lds(g0, g1, z4, z4, 0);
#endif
}

// ---------------------------------------------------------------------------
// Implicit-GEMM convolution on the WMMA pipe. NHWC activations, TDM weights.
//   D[16*NT co, 16 px/wave] += A(W[co][K]) x B(im2col[K][px]), K stepped by 32.
// Block = 256 threads = 8 waves = 128 consecutive pixels of one image row.
// B fragment: per lane-half the 16 K values live in one (r,s) block ->
// 16 contiguous f16 in NHWC = two aligned b128 loads, one bounds check.
// (r,s,c) tracked incrementally: ci >= 32, so c0 advances by 32 with at most
// one carry into rs per step -> no divisions in the hot loop.
// Epilogue: rows j=0..7 are 8 consecutive channels -> one b128 store per half.
// ---------------------------------------------------------------------------
template <int KH, int KW, int NT>
__global__ void __launch_bounds__(256)
conv_wmma_kernel(const _Float16* __restrict__ X, int ci, int xC, int xco,
                 const _Float16* __restrict__ Wt,
                 const float* __restrict__ bias,
                 const float* __restrict__ bnw, const float* __restrict__ bnb,
                 const float* __restrict__ bnm, const float* __restrict__ bnv,
                 const _Float16* __restrict__ Res, int rC, int rco,
                 _Float16* __restrict__ Y, int yC, int yco,
                 int H, int W, int pad, int dil) {
    extern __shared__ char smem_raw[];                 // 2 * NT*16*32 f16
    _Float16* smem = (_Float16*)smem_raw;
    constexpr int KHW  = KH * KW;
    constexpr int ROWS = 16 * NT;
    constexpr int TILE = ROWS * 32;

    const int HWc  = H * W;
    const int lane = threadIdx.x & 31;
    const int wave = threadIdx.x >> 5;
    const int hh   = lane >> 4;                        // lane half
    const int nn   = lane & 15;                        // pixel col / A row m
    const int p    = blockIdx.x * 128 + wave * 16 + nn;
    const int b    = blockIdx.z;
    const int cob0 = blockIdx.y * ROWS;
    const int h    = p / W;
    const int w    = p - h * W;
    const int Kdim = ci * KHW;                         // multiple of 32
    const int nsteps = Kdim / 32;

    const _Float16* __restrict__ Xb = X + (size_t)b * HWc * xC + xco;
    const _Float16* __restrict__ Wb = Wt + (size_t)cob0 * Kdim;

    v8f acc[NT];
    #pragma unroll
    for (int t = 0; t < NT; ++t)
        acc[t] = (v8f){0.f, 0.f, 0.f, 0.f, 0.f, 0.f, 0.f, 0.f};

    if (wave == 0) tdm_load_a_tile(Wb, 0u, Kdim, ROWS);

    // incremental (rs, c) tracking for this lane-half's B run: kk0 = 32*st+16*hh
    int rs0 = 0;
    int c0  = 16 * hh;                                 // 16*hh < 32 <= ci

    for (int st = 0; st < nsteps; ++st) {
        const int k0  = st * 32;
        const int buf = st & 1;
        if (wave == 0) __builtin_amdgcn_s_wait_tensorcnt(0);
        __syncthreads();                               // A tile `buf` published
        if (wave == 0 && st + 1 < nsteps)
            tdm_load_a_tile(Wb + (size_t)(k0 + 32),
                            (unsigned)((buf ^ 1) * TILE * 2), Kdim, ROWS);

        // ---- B fragment: 16 contiguous f16 at one (ih,iw) pixel ----
        const int r  = rs0 / KW;                       // compile-time KW
        const int s  = rs0 - r * KW;
        const int ih = h + dil * r - pad;
        const int iw = w + dil * s - pad;
        v16h bm;
        if ((unsigned)ih < (unsigned)H && (unsigned)iw < (unsigned)W) {
            const _Float16* srcp = Xb + (size_t)(ih * W + iw) * xC + c0;
            v8h lo = *(const v8h*)(srcp);
            v8h hi = *(const v8h*)(srcp + 8);
            #pragma unroll
            for (int i = 0; i < 8; ++i) { bm[i] = lo[i]; bm[8 + i] = hi[i]; }
        } else {
            #pragma unroll
            for (int i = 0; i < 16; ++i) bm[i] = (_Float16)0.f;
        }
        c0 += 32;
        if (c0 >= ci) { c0 -= ci; rs0 += 1; }

        // ---- NT x WMMA, A fragments from LDS (paired 32-bit ds reads) ----
        const _Float16* abase = smem + buf * TILE;
        #pragma unroll
        for (int t = 0; t < NT; ++t) {
            const _Float16* ar = abase + (t * 16 + nn) * 32;
            v16h a;
            #pragma unroll
            for (int i = 0; i < 8; ++i) {
                int kk = 16 * hh + 2 * i;
                a[2 * i]     = ar[kk];
                a[2 * i + 1] = ar[kk + 1];
            }
            acc[t] = __builtin_amdgcn_wmma_f32_16x16x32_f16(
                         false, a, false, bm, (short)0, acc[t], false, false);
        }
        __syncthreads();                               // reads done before overwrite
    }

    // ---- Epilogue: bias -> leaky relu -> BN -> residual -> b128 store ----
    _Float16* Yb = Y + (size_t)b * HWc * yC + (size_t)p * yC + yco;
    const _Float16* Rb = Res ? (Res + (size_t)b * HWc * rC + (size_t)p * rC + rco)
                             : nullptr;
    #pragma unroll
    for (int t = 0; t < NT; ++t) {
        const int cbase = cob0 + t * 16 + 8 * hh;      // 8 consecutive channels
        v8h rv;
        if (Rb) rv = *(const v8h*)(Rb + cbase);
        v8h outv;
        #pragma unroll
        for (int j = 0; j < 8; ++j) {
            int   c = cbase + j;
            float v = acc[t][j] + bias[c];
            v = (v > 0.f) ? v : 0.01f * v;
            if (bnw) {
                float sc = bnw[c] * rsqrtf(bnv[c] + 1e-5f);
                v = (v - bnm[c]) * sc + bnb[c];
            }
            if (Rb) v += (float)rv[j];
            outv[j] = (_Float16)v;
        }
        *(v8h*)(Yb + cbase) = outv;
    }
}

// ---------------------------------------------------------------------------
// KNN front-end: 5x5 / 7x7 range-diff top-9 selection, spherical features,
// two 32x45 matvecs (+ReLU). One thread per (b, pixel). NHWC f16 outputs.
// ---------------------------------------------------------------------------
__global__ void __launch_bounds__(256)
knn_kernel(const float* __restrict__ x, const float* __restrict__ px,
           const float* __restrict__ rw, const float* __restrict__ prw,
           _Float16* __restrict__ kout, _Float16* __restrict__ pkout) {
    int idx = blockIdx.x * blockDim.x + threadIdx.x;
    if (idx >= BN_ * HWF_) return;
    int p = idx % HWF_, b = idx / HWF_;
    int h = p / WF_,   w = p - h * WF_;
    const float* xb  = x  + (size_t)b * CIN_ * HWF_;
    const float* pxb = px + (size_t)b * CIN_ * HWF_;
    float cen = xb[p];

    float neg[25]; int qidx[25];
    {
        int t = 0;
        for (int r = 0; r < 5; ++r)
            for (int s = 0; s < 5; ++s, ++t) {
                int ih = h + r - 2, iw = w + s - 2;
                bool ok = (unsigned)ih < (unsigned)HF_ && (unsigned)iw < (unsigned)WF_;
                float val = ok ? xb[ih * WF_ + iw] : 0.f;
                float d = fabsf(val - cen);
                if (t == 12) d = -1.f;
                neg[t]  = -d;
                qidx[t] = ok ? (ih * WF_ + iw) : -1;
            }
    }
    int sel[9];
    {
        unsigned used = 0u;
        for (int kk = 0; kk < 9; ++kk) {
            float best = -1e30f; int bi = 0;
            for (int j = 0; j < 25; ++j)
                if (!((used >> j) & 1u) && neg[j] > best) { best = neg[j]; bi = j; }
            used |= 1u << bi;
            sel[kk] = bi;
        }
    }
    float g[5][9];
    for (int kk = 0; kk < 9; ++kk) {
        int q = qidx[sel[kk]];
        for (int c = 0; c < 5; ++c) g[c][kk] = (q >= 0) ? xb[(size_t)c * HWF_ + q] : 0.f;
    }
    _Float16* ko = kout + (size_t)idx * 32;
    for (int o = 0; o < 32; ++o) {
        float s = 0.f;
        for (int c = 0; c < 5; ++c)
            for (int kk = 0; kk < 9; ++kk)
                s += rw[o * 45 + c * 9 + kk] * g[c][kk];
        ko[o] = (_Float16)fmaxf(s, 0.f);
    }

    float pneg[49]; int pq[49];
    {
        int t = 0;
        for (int r = 0; r < 7; ++r)
            for (int s = 0; s < 7; ++s, ++t) {
                int ih = h + r - 3, iw = w + s - 3;
                bool ok = (unsigned)ih < (unsigned)HF_ && (unsigned)iw < (unsigned)WF_;
                float val = ok ? pxb[ih * WF_ + iw] : 0.f;
                pneg[t] = -fabsf(val - cen);
                pq[t]   = ok ? (ih * WF_ + iw) : -1;
            }
    }
    int psel[9];
    {
        unsigned long long used = 0ull;
        for (int kk = 0; kk < 9; ++kk) {
            float best = -1e30f; int bi = 0;
            for (int j = 0; j < 49; ++j)
                if (!((used >> j) & 1ull) && pneg[j] > best) { best = pneg[j]; bi = j; }
            used |= 1ull << bi;
            psel[kk] = bi;
        }
    }
    float pg[5][9];
    for (int kk = 0; kk < 9; ++kk) {
        int q = pq[psel[kk]];
        for (int c = 0; c < 5; ++c) pg[c][kk] = (q >= 0) ? pxb[(size_t)c * HWF_ + q] : 0.f;
    }
    float ax = xb[(size_t)1 * HWF_ + p], ay = xb[(size_t)2 * HWF_ + p], az = xb[(size_t)3 * HWF_ + p];
    float pf[45];
    for (int kk = 0; kk < 9; ++kk) {
        float dx = pg[1][kk] - ax, dy = pg[2][kk] - ay, dz = pg[3][kk] - az;
        float xy = dx * dx + dy * dy;
        pf[kk]      = pg[0][kk];
        pf[9 + kk]  = sqrtf(xy + dz * dz);
        pf[18 + kk] = atan2f(sqrtf(xy), dz * dz);
        pf[27 + kk] = atan2f(dy * dy, dx * dx);
        pf[36 + kk] = pg[4][kk];
    }
    _Float16* pko = pkout + (size_t)idx * 32;
    for (int o = 0; o < 32; ++o) {
        float s = 0.f;
        for (int f = 0; f < 45; ++f) s += prw[o * 45 + f] * pf[f];
        pko[o] = (_Float16)fmaxf(s, 0.f);
    }
}

// ---------------------------------------------------------------------------
// 3x3 stride-2 pad-1 average pooling (/9), NHWC f16 -> f16 (c fastest = coalesced)
// ---------------------------------------------------------------------------
__global__ void avgpool_kernel(const _Float16* __restrict__ in,
                               _Float16* __restrict__ out,
                               int C, int H, int W) {
    int H2 = H >> 1, W2 = W >> 1;
    int idx = blockIdx.x * blockDim.x + threadIdx.x;
    int total = BN_ * H2 * W2 * C;
    if (idx >= total) return;
    int c = idx % C; int t = idx / C;
    int j = t % W2;  t /= W2;
    int i = t % H2;  int b = t / H2;
    const _Float16* src = in + (size_t)b * H * W * C;
    float s = 0.f;
    for (int r = 0; r < 3; ++r)
        for (int q = 0; q < 3; ++q) {
            int ih = 2 * i - 1 + r, iw = 2 * j - 1 + q;
            if ((unsigned)ih < (unsigned)H && (unsigned)iw < (unsigned)W)
                s += (float)src[(size_t)(ih * W + iw) * C + c];
        }
    out[idx] = (_Float16)(s * (1.f / 9.f));
}

// ---------------------------------------------------------------------------
// MGA: sa = sigmoid(conv1x1(pre)) * x  (NHWC, per-pixel contiguous 64ch)
// ---------------------------------------------------------------------------
__global__ void mga_fm_sa_kernel(const _Float16* __restrict__ down,
                                 const _Float16* __restrict__ pre,
                                 const float* __restrict__ c2w, const float* __restrict__ c2b,
                                 _Float16* __restrict__ sa, int HWc) {
    int idx = blockIdx.x * blockDim.x + threadIdx.x;
    if (idx >= BN_ * HWc) return;
    const _Float16* pb = pre  + (size_t)idx * 64;
    const _Float16* db = down + (size_t)idx * 64;
    _Float16*       sb = sa   + (size_t)idx * 64;
    float s = c2b[0];
    for (int c = 0; c < 64; ++c) s += c2w[c] * (float)pb[c];
    float fm = 1.f / (1.f + __expf(-s));
    for (int c = 0; c < 64; ++c) sb[c] = (_Float16)(fm * (float)db[c]);
}

// Per-(b,c) channel mean via LDS tree reduction. grid = B*64 blocks. NHWC in.
__global__ void chan_mean_kernel(const _Float16* __restrict__ sa,
                                 float* __restrict__ vmean, int HWc) {
    int bc = blockIdx.x;
    int b = bc >> 6, c = bc & 63;
    const _Float16* src = sa + (size_t)b * HWc * 64 + c;
    float s = 0.f;
    for (int i = threadIdx.x; i < HWc; i += 256) s += (float)src[(size_t)i * 64];
    __shared__ float red[256];
    red[threadIdx.x] = s;
    __syncthreads();
    for (int st = 128; st > 0; st >>= 1) {
        if (threadIdx.x < st) red[threadIdx.x] += red[threadIdx.x + st];
        __syncthreads();
    }
    if (threadIdx.x == 0) vmean[bc] = red[0] / (float)HWc;
}

// v = softmax(conv1x1(mean)) * 64. grid = B blocks of 64 threads.
__global__ void mga_v_kernel(const float* __restrict__ vmean,
                             const float* __restrict__ c1w, const float* __restrict__ c1b,
                             float* __restrict__ vfac) {
    int b = blockIdx.x, c = threadIdx.x;
    float s = c1b[c];
    for (int k = 0; k < 64; ++k) s += c1w[c * 64 + k] * vmean[b * 64 + k];
    __shared__ float sh[64];
    __shared__ float se[64];
    sh[c] = s;
    __syncthreads();
    float mx = sh[0];
    for (int k = 1; k < 64; ++k) mx = fmaxf(mx, sh[k]);
    float e = __expf(s - mx);
    se[c] = e;
    __syncthreads();
    float sum = 0.f;
    for (int k = 0; k < 64; ++k) sum += se[k];
    vfac[b * 64 + c] = (e / sum) * 64.f;
}

// out = sa * v + x   (NHWC)
__global__ void mga_apply_kernel(const _Float16* __restrict__ sa,
                                 const _Float16* __restrict__ down,
                                 const float* __restrict__ vfac,
                                 _Float16* __restrict__ out, int HWc) {
    int idx = blockIdx.x * blockDim.x + threadIdx.x;
    if (idx >= BN_ * HWc) return;
    int b = idx / HWc;
    const _Float16* sb = sa   + (size_t)idx * 64;
    const _Float16* db = down + (size_t)idx * 64;
    _Float16*       ob = out  + (size_t)idx * 64;
    for (int c = 0; c < 64; ++c)
        ob[c] = (_Float16)((float)sb[c] * vfac[b * 64 + c] + (float)db[c]);
}

// ---------------------------------------------------------------------------
// pixel_shuffle(res7_out, r=2) (NHWC 128ch half-res -> 32ch full-res) || skip
// One thread per (b, pixel): writes all 96 channels (64 of them contiguous).
// ---------------------------------------------------------------------------
__global__ void shuffle_concat_kernel(const _Float16* __restrict__ x7,
                                      const _Float16* __restrict__ skip,
                                      _Float16* __restrict__ up) {
    int idx = blockIdx.x * blockDim.x + threadIdx.x;
    if (idx >= BN_ * HWF_) return;
    int p = idx % HWF_, b = idx / HWF_;
    int h = p / WF_,   w = p - h * WF_;
    int r = h & 1, s = w & 1, i = h >> 1, j = w >> 1;
    const _Float16* x7p = x7 + ((size_t)b * HWH_ + (size_t)(i * WH_ + j)) * 128;
    const _Float16* skp = skip + (size_t)idx * 64;
    _Float16*       upp = up   + (size_t)idx * 96;
    int cbase = r * 2 + s;
    for (int c = 0; c < 32; ++c) upp[c] = x7p[c * 4 + cbase];
    for (int c = 0; c < 64; ++c) upp[32 + c] = skp[c];
}

// ---------------------------------------------------------------------------
// logits (1x1 conv 32->3) + class softmax; NHWC in, NCHW f32 out (reference)
// ---------------------------------------------------------------------------
__global__ void logits_softmax_kernel(const _Float16* __restrict__ up,
                                      const float* __restrict__ lw,
                                      const float* __restrict__ lb,
                                      float* __restrict__ out) {
    int idx = blockIdx.x * blockDim.x + threadIdx.x;
    if (idx >= BN_ * HWF_) return;
    int p = idx % HWF_, b = idx / HWF_;
    const _Float16* ub = up + (size_t)idx * 32;
    float l[3];
    for (int cls = 0; cls < 3; ++cls) {
        float s = lb[cls];
        for (int c = 0; c < 32; ++c) s += lw[cls * 32 + c] * (float)ub[c];
        l[cls] = s;
    }
    float mx = fmaxf(l[0], fmaxf(l[1], l[2]));
    float e0 = __expf(l[0] - mx), e1 = __expf(l[1] - mx), e2 = __expf(l[2] - mx);
    float inv = 1.f / (e0 + e1 + e2);
    out[((size_t)b * 3 + 0) * HWF_ + p] = e0 * inv;
    out[((size_t)b * 3 + 1) * HWF_ + p] = e1 * inv;
    out[((size_t)b * 3 + 2) * HWF_ + p] = e2 * inv;
}

// ---------------------------------------------------------------------------
// Host side
// ---------------------------------------------------------------------------
// d_in layout (pytree order): 0:x, 1:pre_x, then params by sorted keys:
//   2:logb 3:logw 4..7:mga{c1b,c1w,c2b,c2w} 8:pre_range_weight 9:range_weight
//   10..35:res1  36..61:res7  62..87:res_pre  88..111:up6
#define R1_ 10
#define R7_ 36
#define RP_ 62
#define U6_ 88

static void launch_conv(hipStream_t st,
                        const _Float16* X, int ci, int xC, int xco,
                        const _Float16* Wt, const float* bias,
                        const float* bnw, const float* bnb,
                        const float* bnm, const float* bnv,
                        const _Float16* Res, int rC, int rco,
                        _Float16* Y, int co, int yC, int yco,
                        int H, int W, int kh, int kw, int pad, int dil) {
    const int NT = (co % 64 == 0) ? 4 : 2;
    dim3 grid((unsigned)(H * W / 128), (unsigned)(co / (16 * NT)), (unsigned)BN_);
    size_t shmem = (size_t)2 * NT * 16 * 32 * sizeof(_Float16);
    #define ARGS X, ci, xC, xco, Wt, bias, bnw, bnb, bnm, bnv, Res, rC, rco, \
                 Y, yC, yco, H, W, pad, dil
    if (kh == 3) {
        if (NT == 4) conv_wmma_kernel<3, 3, 4><<<grid, 256, shmem, st>>>(ARGS);
        else         conv_wmma_kernel<3, 3, 2><<<grid, 256, shmem, st>>>(ARGS);
    } else if (kh == 2) {
        if (NT == 4) conv_wmma_kernel<2, 2, 4><<<grid, 256, shmem, st>>>(ARGS);
        else         conv_wmma_kernel<2, 2, 2><<<grid, 256, shmem, st>>>(ARGS);
    } else {
        if (NT == 4) conv_wmma_kernel<1, 1, 4><<<grid, 256, shmem, st>>>(ARGS);
        else         conv_wmma_kernel<1, 1, 2><<<grid, 256, shmem, st>>>(ARGS);
    }
    #undef ARGS
}

extern "C" void kernel_launch(void* const* d_in, const int* in_sizes, int n_in,
                              void* d_out, int out_size, void* d_ws, size_t ws_size,
                              hipStream_t stream) {
    (void)in_sizes; (void)n_in; (void)out_size; (void)ws_size;
    const float* x   = (const float*)d_in[0];
    const float* pre = (const float*)d_in[1];
    auto F = [&](int i) -> const float* { return (const float*)d_in[i]; };

    // ---- workspace bump allocator (256B aligned) with explicit aliasing ----
    char*  ws  = (char*)d_ws;
    size_t off = 0;
    auto alloc = [&](size_t bytes) -> char* {
        char* r = ws + off;
        off = (off + bytes + 255) & ~(size_t)255;
        return r;
    };

    _Float16* w16    = (_Float16*)alloc((size_t)2 << 20);                 // packed f16 weights
    _Float16* k16    = (_Float16*)alloc((size_t)BN_ * HWF_ * 32 * 2);     // NHWC
    _Float16* pk16   = (_Float16*)alloc((size_t)BN_ * HWF_ * 32 * 2);
    char*     BIG    = alloc((size_t)BN_ * HWF_ * 192 * 2);               // 201MB shared region
    _Float16* sc64   = (_Float16*)alloc((size_t)BN_ * HWF_ * 64 * 2);
    _Float16* skip   = (_Float16*)alloc((size_t)BN_ * HWF_ * 64 * 2);
    _Float16* preout = (_Float16*)alloc((size_t)BN_ * HWF_ * 64 * 2);
    _Float16* down   = (_Float16*)alloc((size_t)BN_ * HWH_ * 64 * 2);
    _Float16* predwn = (_Float16*)alloc((size_t)BN_ * HWH_ * 64 * 2);
    _Float16* sa     = (_Float16*)alloc((size_t)BN_ * HWH_ * 64 * 2);
    _Float16* mgaout = (_Float16*)alloc((size_t)BN_ * HWH_ * 64 * 2);
    _Float16* sc7    = (_Float16*)alloc((size_t)BN_ * HWH_ * 128 * 2);
    _Float16* out7   = (_Float16*)alloc((size_t)BN_ * HWH_ * 128 * 2);
    float*    vmean  = (float*)alloc(BN_ * 64 * sizeof(float));
    float*    vfac   = (float*)alloc(BN_ * 64 * sizeof(float));

    // aliases inside BIG (lifetimes do not overlap):
    _Float16* cat192 = (_Float16*)BIG;                                    // res1/res_pre a1|a2|a3 (NHWC 192)
    _Float16* cat384 = (_Float16*)BIG;                                    // res7 a1|a2|a3 (NHWC 384, half-res)
    _Float16* upcat  = (_Float16*)BIG;                                    // NHWC 96 up-concat (after res7)
    _Float16* ecat   = (_Float16*)(BIG + (size_t)BN_ * HWF_ * 96 * 2);    // NHWC 96 e1|e2|e3
    _Float16* upout  = preout;                                            // preout free after pooling

    // ---- repack all conv weights: f32 OIHW -> f16 [co][(r,s,c)] ----
    size_t wcur = 0;
    auto pack = [&](int pidx, int co, int ci, int khw) -> _Float16* {
        int n = co * ci * khw;
        _Float16* dst = w16 + wcur;
        wcur += (size_t)n;
        pack_weight_kernel<<<(n + 255) / 256, 256, 0, stream>>>(F(pidx), dst, ci, khw, n);
        return dst;
    };
    _Float16* r1w1 = pack(R1_ + 17, 64, 32, 1);
    _Float16* r1w2 = pack(R1_ + 19, 64, 32, 9);
    _Float16* r1w3 = pack(R1_ + 21, 64, 64, 9);
    _Float16* r1w4 = pack(R1_ + 23, 64, 64, 4);
    _Float16* r1w5 = pack(R1_ + 25, 64, 192, 1);
    _Float16* rpw1 = pack(RP_ + 17, 64, 32, 1);
    _Float16* rpw2 = pack(RP_ + 19, 64, 32, 9);
    _Float16* rpw3 = pack(RP_ + 21, 64, 64, 9);
    _Float16* rpw4 = pack(RP_ + 23, 64, 64, 4);
    _Float16* rpw5 = pack(RP_ + 25, 64, 192, 1);
    _Float16* r7w1 = pack(R7_ + 17, 128, 64, 1);
    _Float16* r7w2 = pack(R7_ + 19, 128, 64, 9);
    _Float16* r7w3 = pack(R7_ + 21, 128, 128, 9);
    _Float16* r7w4 = pack(R7_ + 23, 128, 128, 4);
    _Float16* r7w5 = pack(R7_ + 25, 128, 384, 1);
    _Float16* u6w1 = pack(U6_ + 17, 32, 96, 9);
    _Float16* u6w2 = pack(U6_ + 19, 32, 32, 9);
    _Float16* u6w3 = pack(U6_ + 21, 32, 32, 4);
    _Float16* u6w4 = pack(U6_ + 23, 32, 96, 1);

    // ---- KNN front-end (outputs NHWC f16) ----
    knn_kernel<<<(BN_ * HWF_ + 255) / 256, 256, 0, stream>>>(
        x, pre, F(9), F(8), k16, pk16);

    // ---- resblock helper (all NHWC) ----
    auto resblock = [&](int R, const _Float16* xin, int ci, int co,
                        const _Float16* w1, const _Float16* w2, const _Float16* w3,
                        const _Float16* w4, const _Float16* w5,
                        _Float16* cat, _Float16* sc, _Float16* outb,
                        int H, int W) {
        // sc = lrelu(conv1x1)
        launch_conv(stream, xin, ci, ci, 0, w1, F(R + 16),
                    nullptr, nullptr, nullptr, nullptr, nullptr, 0, 0,
                    sc, co, co, 0, H, W, 1, 1, 0, 1);
        // a1 = bn1(lrelu(conv3x3 pad1))
        launch_conv(stream, xin, ci, ci, 0, w2, F(R + 18),
                    F(R + 3), F(R + 0), F(R + 1), F(R + 2), nullptr, 0, 0,
                    cat, co, 3 * co, 0, H, W, 3, 3, 1, 1);
        // a2 = bn2(lrelu(conv3x3 pad2 dil2))
        launch_conv(stream, cat, co, 3 * co, 0, w3, F(R + 20),
                    F(R + 7), F(R + 4), F(R + 5), F(R + 6), nullptr, 0, 0,
                    cat, co, 3 * co, co, H, W, 3, 3, 2, 2);
        // a3 = bn3(lrelu(conv2x2 pad1 dil2))
        launch_conv(stream, cat, co, 3 * co, co, w4, F(R + 22),
                    F(R + 11), F(R + 8), F(R + 9), F(R + 10), nullptr, 0, 0,
                    cat, co, 3 * co, 2 * co, H, W, 2, 2, 1, 2);
        // out = bn4(lrelu(conv1x1(cat3))) + sc
        launch_conv(stream, cat, 3 * co, 3 * co, 0, w5, F(R + 24),
                    F(R + 15), F(R + 12), F(R + 13), F(R + 14),
                    sc, co, 0,
                    outb, co, co, 0, H, W, 1, 1, 0, 1);
    };

    // ---- res1 (32 -> 64, full res) + pool ----
    resblock(R1_, k16, 32, 64, r1w1, r1w2, r1w3, r1w4, r1w5, cat192, sc64, skip, HF_, WF_);
    avgpool_kernel<<<(BN_ * HWH_ * 64 + 255) / 256, 256, 0, stream>>>(skip, down, 64, HF_, WF_);

    // ---- res_pre (32 -> 64, full res) + pool ----
    resblock(RP_, pk16, 32, 64, rpw1, rpw2, rpw3, rpw4, rpw5, cat192, sc64, preout, HF_, WF_);
    avgpool_kernel<<<(BN_ * HWH_ * 64 + 255) / 256, 256, 0, stream>>>(preout, predwn, 64, HF_, WF_);

    // ---- MGA ----
    mga_fm_sa_kernel<<<(BN_ * HWH_ + 255) / 256, 256, 0, stream>>>(
        down, predwn, F(7), F(6), sa, HWH_);
    chan_mean_kernel<<<BN_ * 64, 256, 0, stream>>>(sa, vmean, HWH_);
    mga_v_kernel<<<BN_, 64, 0, stream>>>(vmean, F(5), F(4), vfac);
    mga_apply_kernel<<<(BN_ * HWH_ + 255) / 256, 256, 0, stream>>>(
        sa, down, vfac, mgaout, HWH_);

    // ---- res7 (64 -> 128, half res, no pooling) ----
    resblock(R7_, mgaout, 64, 128, r7w1, r7w2, r7w3, r7w4, r7w5, cat384, sc7, out7, HH_, WH_);

    // ---- upblock: pixel_shuffle(out7) || skip -> NHWC 96, then conv chain ----
    shuffle_concat_kernel<<<(BN_ * HWF_ + 255) / 256, 256, 0, stream>>>(
        out7, skip, upcat);
    launch_conv(stream, upcat, 96, 96, 0, u6w1, F(U6_ + 16),
                F(U6_ + 3), F(U6_ + 0), F(U6_ + 1), F(U6_ + 2), nullptr, 0, 0,
                ecat, 32, 96, 0, HF_, WF_, 3, 3, 1, 1);
    launch_conv(stream, ecat, 32, 96, 0, u6w2, F(U6_ + 18),
                F(U6_ + 7), F(U6_ + 4), F(U6_ + 5), F(U6_ + 6), nullptr, 0, 0,
                ecat, 32, 96, 32, HF_, WF_, 3, 3, 2, 2);
    launch_conv(stream, ecat, 32, 96, 32, u6w3, F(U6_ + 20),
                F(U6_ + 11), F(U6_ + 8), F(U6_ + 9), F(U6_ + 10), nullptr, 0, 0,
                ecat, 32, 96, 64, HF_, WF_, 2, 2, 1, 2);
    launch_conv(stream, ecat, 96, 96, 0, u6w4, F(U6_ + 22),
                F(U6_ + 15), F(U6_ + 12), F(U6_ + 13), F(U6_ + 14), nullptr, 0, 0,
                upout, 32, 32, 0, HF_, WF_, 1, 1, 0, 1);

    // ---- logits + softmax -> d_out (f32, NCHW as reference) ----
    logits_softmax_kernel<<<(BN_ * HWF_ + 255) / 256, 256, 0, stream>>>(
        upout, F(3), F(2), (float*)d_out);
}
